// E88FusedLayer_19052474925241
// MI455X (gfx1250) — matrix operand order repaired and checked by reading, not compiled, not run on hardware
//
#include <hip/hip_runtime.h>
#include <hip/hip_bf16.h>
#include <stdint.h>

typedef __attribute__((ext_vector_type(16))) __bf16 v16bf;
typedef __attribute__((ext_vector_type(8)))  float   v8f;
typedef __attribute__((ext_vector_type(4)))  int     v4i;

namespace cfg {
constexpr int  B = 8, T = 2048, D = 2048, H = 32, Ns = 32, Vd = 32;
constexpr int  KEY_DIM   = H * Ns;                    // 1024
constexpr int  VALUE_DIM = H * Vd;                    // 1024
constexpr int  NCAT = 2 * KEY_DIM + H + VALUE_DIM;    // 3104: [q|k|v|alpha|gate]
constexpr long M = (long)B * T;                       // 16384 rows
}

// ---------------------------------------------------------------------------
// gfx1250 async global->LDS staging (ASYNCcnt path). Guarded so the file
// still compiles (sync fallback) if the toolchain lacks the builtins.
// Builtin signature (from clang diagnostic): (v4i AS1*, v4i AS3*, Imm, Imm).
// ---------------------------------------------------------------------------
#if defined(__gfx1250__) &&                                              \
    __has_builtin(__builtin_amdgcn_global_load_async_to_lds_b128) &&     \
    __has_builtin(__builtin_amdgcn_s_wait_asynccnt)
#define USE_ASYNC_LDS 1
typedef __attribute__((address_space(1))) v4i* gp128;
typedef __attribute__((address_space(3))) v4i* lp128;
__device__ __forceinline__ void async_copy_b128(const void* g, void* l) {
  // generic->addrspace via integer casts (low 32 bits of generic LDS ptr ==
  // LDS offset) so this compiles regardless of addrspace-cast rules.
  __builtin_amdgcn_global_load_async_to_lds_b128(
      (gp128)(uintptr_t)g, (lp128)(uint32_t)(uintptr_t)l, 0, 0);
}
#else
#define USE_ASYNC_LDS 0
#endif

// ---------------------------------------------------------------------------
// fp32 -> bf16 conversion (vectorized, 4 elems/thread)
// ---------------------------------------------------------------------------
__global__ void cvt_f32_bf16(const float* __restrict__ in, __bf16* __restrict__ out,
                             size_t n) {
  size_t i = ((size_t)blockIdx.x * blockDim.x + threadIdx.x) * 4;
  if (i + 3 < n) {
    float4 v = *(const float4*)(in + i);
    union { __bf16 h[4]; unsigned long long u; } p;
    p.h[0] = (__bf16)v.x; p.h[1] = (__bf16)v.y;
    p.h[2] = (__bf16)v.z; p.h[3] = (__bf16)v.w;
    *(unsigned long long*)(out + i) = p.u;
  } else {
    for (; i < n; ++i) out[i] = (__bf16)in[i];
  }
}

// Pack [W_qkv | W_a | W_g] into N-major (transposed) bf16 Wt[NCAT][D].
__global__ void pack_wcat(const float* __restrict__ Wqkv, const float* __restrict__ Wa,
                          const float* __restrict__ Wg, __bf16* __restrict__ Wt) {
  long id = (long)blockIdx.x * blockDim.x + threadIdx.x;
  long total = (long)cfg::NCAT * cfg::D;
  if (id >= total) return;
  int k = (int)(id / cfg::NCAT);
  int j = (int)(id % cfg::NCAT);
  float v;
  if (j < 3072)      v = Wqkv[(long)k * 3072 + j];
  else if (j < 3104) v = Wa  [(long)k * 32   + (j - 3072)];
  else               v = Wg  [(long)k * 1024 + (j - 3104)];
  Wt[(long)j * cfg::D + k] = (__bf16)v;
}

// W_o [VALUE_DIM, D] fp32 -> Wt_o [D][VALUE_DIM] bf16
__global__ void pack_wo(const float* __restrict__ Wo, __bf16* __restrict__ Wt) {
  long id = (long)blockIdx.x * blockDim.x + threadIdx.x;
  long total = (long)cfg::VALUE_DIM * cfg::D;
  if (id >= total) return;
  int k = (int)(id / cfg::D);
  int n = (int)(id % cfg::D);
  Wt[(long)n * cfg::VALUE_DIM + k] = (__bf16)Wo[(long)k * cfg::D + n];
}

// ---------------------------------------------------------------------------
// bf16 WMMA GEMM:  C[M][N] (f32) = A[M][K] (bf16 row-major) * Bt[N][K]^T
// Block: 256 threads = 8 waves; block tile 128x64; BK = 32 per stage.
// Wave grid 4(M) x 2(N); each wave owns a 32x32 register tile:
//   2 A frags x 2 B frags -> 4 v_wmma per stage.
// Async path: double-buffered LDS filled by global_load_async_to_lds_b128,
// 3 b128 copies per thread per stage, s_wait_asynccnt pipelining.
// OOB tile rows are clamped (not zero-filled): they only feed accumulator
// columns/rows >= N/M which are never stored.
// ---------------------------------------------------------------------------
union Frag { uint4 u[2]; v16bf v; };

__global__ __launch_bounds__(256) void gemm_bf16_wmma(
    const __bf16* __restrict__ A, const __bf16* __restrict__ Bt,
    float* __restrict__ C, int M, int N, int K) {
  __shared__ __align__(16) __bf16 As[2][128][32];   // 2 x 8 KB
  __shared__ __align__(16) __bf16 Bs[2][64][32];    // 2 x 4 KB

  const int tid  = threadIdx.x;
  const int lane = tid & 31;
  const int wave = tid >> 5;
  const int wm   = (wave & 3) << 5;     // wave M offset: 0,32,64,96
  const int wn   = (wave >> 2) << 5;    // wave N offset: 0,32
  const long m0  = (long)blockIdx.x * 128;
  const long n0  = (long)blockIdx.y * 64;

  // cooperative copy mapping
  const int arow = tid >> 1;            // 0..127
  const int acol = (tid & 1) << 4;      // 0,16  (2 x b128)
  const int brow = tid >> 2;            // 0..63
  const int bcol = (tid & 3) << 3;      // 0,8,16,24 (1 x b128)
  const long gmr = (m0 + arow < M) ? (m0 + arow) : (M - 1);   // clamped
  const long gnr = (n0 + brow < N) ? (n0 + brow) : (N - 1);
  const __bf16* gA = A  + gmr * K;
  const __bf16* gB = Bt + gnr * K;

  // fragment mapping (ISA 7.12.2, 16-bit A 16x32 / B 32x16)
  const int mr    = lane & 15;
  const int koffA = (lane < 16) ? 0 : 8;    // A: K {0..7,16..23}/{8..15,24..31}
  const int koffB = (lane < 16) ? 0 : 8;    // B: K {0..7,16..23}/{8..15,24..31}

  v8f acc00 = {0.f,0.f,0.f,0.f,0.f,0.f,0.f,0.f};
  v8f acc01 = acc00, acc10 = acc00, acc11 = acc00;

  const int nstage = K >> 5;

#if USE_ASYNC_LDS
  // prologue: stage 0 -> buffer 0
  async_copy_b128(gA + acol,      &As[0][arow][acol]);
  async_copy_b128(gA + acol + 8,  &As[0][arow][acol + 8]);
  async_copy_b128(gB + bcol,      &Bs[0][brow][bcol]);
  for (int s = 0; s < nstage; ++s) {
    const int cur = s & 1;
    if (s + 1 < nstage) {
      const int nxt = cur ^ 1;
      const int k1 = (s + 1) << 5;
      async_copy_b128(gA + k1 + acol,     &As[nxt][arow][acol]);
      async_copy_b128(gA + k1 + acol + 8, &As[nxt][arow][acol + 8]);
      async_copy_b128(gB + k1 + bcol,     &Bs[nxt][brow][bcol]);
      __builtin_amdgcn_s_wait_asynccnt(3);   // stage s landed (in-order)
    } else {
      __builtin_amdgcn_s_wait_asynccnt(0);
    }
    __syncthreads();                         // buffer cur visible to all waves

    Frag a0, a1, b0, b1;
    a0.u[0] = *(const uint4*)&As[cur][wm + mr][koffA];
    a0.u[1] = *(const uint4*)&As[cur][wm + mr][koffA + 16];
    a1.u[0] = *(const uint4*)&As[cur][wm + 16 + mr][koffA];
    a1.u[1] = *(const uint4*)&As[cur][wm + 16 + mr][koffA + 16];
    b0.u[0] = *(const uint4*)&Bs[cur][wn + mr][koffB];
    b0.u[1] = *(const uint4*)&Bs[cur][wn + mr][koffB + 16];
    b1.u[0] = *(const uint4*)&Bs[cur][wn + 16 + mr][koffB];
    b1.u[1] = *(const uint4*)&Bs[cur][wn + 16 + mr][koffB + 16];

    acc00 = __builtin_amdgcn_wmma_f32_16x16x32_bf16(false, a0.v, false, b0.v,
                                                    (short)0, acc00, false, false);
    acc01 = __builtin_amdgcn_wmma_f32_16x16x32_bf16(false, a0.v, false, b1.v,
                                                    (short)0, acc01, false, false);
    acc11 = __builtin_amdgcn_wmma_f32_16x16x32_bf16(false, a1.v, false, b1.v,
                                                    (short)0, acc11, false, false);
    acc10 = __builtin_amdgcn_wmma_f32_16x16x32_bf16(false, a1.v, false, b0.v,
                                                    (short)0, acc10, false, false);
    __syncthreads();                         // reads done before cur is refilled
  }
#else
  for (int s = 0; s < nstage; ++s) {
    const int k0 = s << 5;
    uint4 va0 = *(const uint4*)(gA + k0 + acol);
    uint4 va1 = *(const uint4*)(gA + k0 + acol + 8);
    uint4 vb0 = *(const uint4*)(gB + k0 + bcol);
    if (k0 + 32 < K) {
      __builtin_prefetch(gA + k0 + 32 + acol, 0, 1);
      __builtin_prefetch(gB + k0 + 32 + bcol, 0, 1);
    }
    __syncthreads();
    *(uint4*)&As[0][arow][acol]     = va0;
    *(uint4*)&As[0][arow][acol + 8] = va1;
    *(uint4*)&Bs[0][brow][bcol]     = vb0;
    __syncthreads();

    Frag a0, a1, b0, b1;
    a0.u[0] = *(const uint4*)&As[0][wm + mr][koffA];
    a0.u[1] = *(const uint4*)&As[0][wm + mr][koffA + 16];
    a1.u[0] = *(const uint4*)&As[0][wm + 16 + mr][koffA];
    a1.u[1] = *(const uint4*)&As[0][wm + 16 + mr][koffA + 16];
    b0.u[0] = *(const uint4*)&Bs[0][wn + mr][koffB];
    b0.u[1] = *(const uint4*)&Bs[0][wn + mr][koffB + 16];
    b1.u[0] = *(const uint4*)&Bs[0][wn + 16 + mr][koffB];
    b1.u[1] = *(const uint4*)&Bs[0][wn + 16 + mr][koffB + 16];

    acc00 = __builtin_amdgcn_wmma_f32_16x16x32_bf16(false, a0.v, false, b0.v,
                                                    (short)0, acc00, false, false);
    acc01 = __builtin_amdgcn_wmma_f32_16x16x32_bf16(false, a0.v, false, b1.v,
                                                    (short)0, acc01, false, false);
    acc11 = __builtin_amdgcn_wmma_f32_16x16x32_bf16(false, a1.v, false, b1.v,
                                                    (short)0, acc11, false, false);
    acc10 = __builtin_amdgcn_wmma_f32_16x16x32_bf16(false, a1.v, false, b0.v,
                                                    (short)0, acc10, false, false);
  }
#endif

  // C/D layout: VGPR r -> (lanes 0-15: M=r, N=lane) (lanes 16-31: M=r+8, N=lane-16)
  const int cl = lane & 15;
  const int rb = (lane < 16) ? 0 : 8;
  const long col0 = n0 + wn + cl;
  const long col1 = col0 + 16;
#pragma unroll
  for (int r = 0; r < 8; ++r) {
    const long row0 = m0 + wm + rb + r;
    const long row1 = row0 + 16;
    if (row0 < M) {
      if (col0 < N) C[row0 * N + col0] = acc00[r];
      if (col1 < N) C[row0 * N + col1] = acc01[r];
    }
    if (row1 < M) {
      if (col0 < N) C[row1 * N + col0] = acc10[r];
      if (col1 < N) C[row1 * N + col1] = acc11[r];
    }
  }
}

// ---------------------------------------------------------------------------
// Fused recurrent scan: one wave32 per (b,h). Lane = V column, S[n] in regs.
// ---------------------------------------------------------------------------
__global__ __launch_bounds__(32) void e88_scan(
    const float* __restrict__ qkvag, const float* __restrict__ A_log,
    const float* __restrict__ dt_bias, const float* __restrict__ S0,
    __bf16* __restrict__ attn_bf, float* __restrict__ S_final) {
  const int bh   = blockIdx.x;
  const int b    = bh >> 5;
  const int h    = bh & 31;
  const int lane = threadIdx.x;

  float S[32];
#pragma unroll
  for (int n = 0; n < 32; ++n)
    S[n] = S0[(((long)b * 32 + h) * 32 + n) * 32 + lane];

  const float beta = __expf(A_log[h]);
  const float dtb  = dt_bias[h];

  const float* row  = qkvag + (long)b * cfg::T * cfg::NCAT;
  __bf16*      orow = attn_bf + (long)b * cfg::T * cfg::VALUE_DIM + h * 32 + lane;

  for (int t = 0; t < cfg::T; ++t, row += cfg::NCAT, orow += cfg::VALUE_DIM) {
    float qv = row[h * 32 + lane];
    float kv = row[1024 + h * 32 + lane];
    float vv = row[2048 + h * 32 + lane];
    float al = row[3072 + h];
    float gv = row[3104 + h * 32 + lane];

    float qs = qv * qv, ks = kv * kv;
#pragma unroll
    for (int off = 16; off >= 1; off >>= 1) {
      qs += __shfl_xor(qs, off, 32);
      ks += __shfl_xor(ks, off, 32);
    }
    qv = qv / fmaxf(sqrtf(qs), 1e-6f);
    kv = kv / fmaxf(sqrtf(ks), 1e-6f);

    float xsp = al + dtb;
    float sp  = (xsp > 20.f) ? xsp : log1pf(__expf(xsp));
    float dcy = __expf(-beta * sp);

    float retr = 0.f;
#pragma unroll
    for (int n = 0; n < 32; ++n)
      retr = fmaf(S[n], __shfl(kv, n, 32), retr);
    float delta = vv - retr;

#pragma unroll
    for (int n = 0; n < 32; ++n)
      S[n] = tanhf(fmaf(__shfl(kv, n, 32), delta, dcy * S[n]));

    float sq = 0.f;
#pragma unroll
    for (int n = 0; n < 32; ++n)
      sq = fmaf(S[n], __shfl(qv, n, 32), sq);

    float sig = 1.f / (1.f + __expf(-gv));
    *orow = (__bf16)(sq * gv * sig);
  }

#pragma unroll
  for (int n = 0; n < 32; ++n)
    S_final[(((long)b * 32 + h) * 32 + n) * 32 + lane] = S[n];
}

// ---------------------------------------------------------------------------
// Host orchestration
// ---------------------------------------------------------------------------
extern "C" void kernel_launch(void* const* d_in, const int* in_sizes, int n_in,
                              void* d_out, int out_size, void* d_ws, size_t ws_size,
                              hipStream_t stream) {
  const float* x       = (const float*)d_in[0];
  const float* W_qkv   = (const float*)d_in[1];
  const float* W_a     = (const float*)d_in[2];
  const float* W_g     = (const float*)d_in[3];
  const float* A_log   = (const float*)d_in[4];
  const float* dt_bias = (const float*)d_in[5];
  const float* W_o     = (const float*)d_in[6];
  const float* S0      = (const float*)d_in[7];
  float* out = (float*)d_out;

  // workspace layout (256B-aligned regions)
  char* ws = (char*)d_ws;
  __bf16* x_bf   = (__bf16*)(ws);                 // 16384*2048*2  = 67108864
  __bf16* wcat_t = (__bf16*)(ws + 67108864);      // 4128*2048*2   = 16908288
  __bf16* wo_t   = (__bf16*)(ws + 84017152);      // 2048*1024*2   =  4194304
  float*  qkvag  = (float* )(ws + 88211456);      // 16384*4128*4  = 270532608
  __bf16* attn   = (__bf16*)(ws + 358744064);     // 16384*1024*2  = 33554432

  const size_t nx = (size_t)cfg::M * cfg::D;      // 33554432
  cvt_f32_bf16<<<(unsigned)((nx / 4 + 255) / 256), 256, 0, stream>>>(x, x_bf, nx);

  const long nw = (long)cfg::NCAT * cfg::D;
  pack_wcat<<<(unsigned)((nw + 255) / 256), 256, 0, stream>>>(W_qkv, W_a, W_g, wcat_t);

  const long nwo = (long)cfg::VALUE_DIM * cfg::D;
  pack_wo<<<(unsigned)((nwo + 255) / 256), 256, 0, stream>>>(W_o, wo_t);

  // GEMM 1: qkvag[M][NCAT] = x_bf[M][D] * wcat_t^T   (block tile 128x64)
  dim3 g1((unsigned)(cfg::M / 128), (unsigned)((cfg::NCAT + 63) / 64));
  gemm_bf16_wmma<<<g1, 256, 0, stream>>>(x_bf, wcat_t, qkvag,
                                         (int)cfg::M, cfg::NCAT, cfg::D);

  // recurrent scan: S_final goes right after out (B*T*D floats) in d_out
  float* S_final = out + (long)cfg::B * cfg::T * cfg::D;
  e88_scan<<<cfg::B * cfg::H, 32, 0, stream>>>(qkvag, A_log, dt_bias, S0,
                                               attn, S_final);

  // GEMM 2: out[M][D] = attn[M][VALUE_DIM] * wo_t^T
  dim3 g2((unsigned)(cfg::M / 128), (unsigned)(cfg::D / 64));
  gemm_bf16_wmma<<<g2, 256, 0, stream>>>(attn, wo_t, out,
                                         (int)cfg::M, cfg::D, cfg::VALUE_DIM);
}
